// AtomAttentionDecoder_40501541601324
// MI455X (gfx1250) — matrix operand order repaired and checked by reading, not compiled, not run on hardware
//
#include <hip/hip_runtime.h>
#include <hip/hip_bf16.h>

typedef __attribute__((ext_vector_type(16))) __bf16 v16bf;
typedef __attribute__((ext_vector_type(8)))  __bf16 v8bf;
typedef __attribute__((ext_vector_type(8)))  float  v8f;
typedef __attribute__((ext_vector_type(4)))  unsigned int u32x4;
typedef __attribute__((ext_vector_type(8)))  int  i32x8;
typedef __attribute__((ext_vector_type(4)))  int  i32x4;

#define DEVI __device__ __forceinline__

// problem sizes
#define BB    2
#define NTOK  512
#define NATOM 16384
#define CT    384
#define CA    128
#define CP    16
#define CS    384
#define NQW   32
#define NKW   128
#define NH    4
#define NW    512
#define NBLK  3
#define MROWS (BB*NATOM)   /* 32768 */

#if __has_builtin(__builtin_amdgcn_tensor_load_to_lds)
#define TDM_OK 1
#else
#define TDM_OK 0
#endif

DEVI __bf16 f2bf(float x) { return (__bf16)x; }

DEVI float wave_sum(float v) {
  #pragma unroll
  for (int m = 16; m; m >>= 1) v += __shfl_xor(v, m, 32);
  return v;
}
DEVI float wave_max(float v) {
  #pragma unroll
  for (int m = 16; m; m >>= 1) v = fmaxf(v, __shfl_xor(v, m, 32));
  return v;
}
DEVI float block_sum_128(float v, float* sbuf) {
  v = wave_sum(v);
  int wv = threadIdx.x >> 5;
  if ((threadIdx.x & 31) == 0) sbuf[wv] = v;
  __syncthreads();
  float r = sbuf[0] + sbuf[1] + sbuf[2] + sbuf[3];
  __syncthreads();
  return r;
}

// 16-bf16 fragment = two 16B LDS chunks (2x ds_load_b128)
DEVI v16bf frag16(const __bf16* p) {
  v8bf a0 = *(const v8bf*)p;
  v8bf a1 = *(const v8bf*)(p + 8);
  return __builtin_shufflevector(a0, a1, 0,1,2,3,4,5,6,7,8,9,10,11,12,13,14,15);
}
DEVI v16bf frag16_gap(const __bf16* p) {       // two 8-elem chunks 16 apart
  v8bf a0 = *(const v8bf*)p;
  v8bf a1 = *(const v8bf*)(p + 16);
  return __builtin_shufflevector(a0, a1, 0,1,2,3,4,5,6,7,8,9,10,11,12,13,14,15);
}
DEVI v8f bwmma(v16bf a, v16bf b, v8f c) {
  return __builtin_amdgcn_wmma_f32_16x16x32_bf16(false, a, false, b,
                                                 (short)0, c, false, false);
}
// 8-element source loads (bf16 copy or f32 convert)
DEVI v8bf load8(const __bf16* p) { return *(const v8bf*)p; }
DEVI v8bf load8(const float* p) {
  v8f f = *(const v8f*)p;
  v8bf r;
  #pragma unroll
  for (int i = 0; i < 8; ++i) r[i] = f2bf(f[i]);
  return r;
}

#if TDM_OK
// TDM: 2D bf16 tile global->LDS, row padded 256B -> 272B (stride 136 elems)
DEVI void tdm_load_2d_bf16(const __bf16* gsrc, unsigned lds_off,
                           int rows, int cols, int src_stride_elems)
{
  unsigned long long ga = (unsigned long long)(uintptr_t)gsrc;
  u32x4 g0;
  g0[0] = 1u;
  g0[1] = lds_off;
  g0[2] = (unsigned)(ga & 0xFFFFFFFFull);
  g0[3] = (unsigned)((ga >> 32) & 0x01FFFFFFull) | 0x80000000u;
  i32x8 g1;
  g1[0] = (1 << 16) | (1 << 20) | (5 << 22) | (3 << 25);
  g1[1] = (cols & 0xFFFF) << 16;
  g1[2] = ((cols >> 16) & 0xFFFF) | ((rows & 0xFFFF) << 16);
  g1[3] = ((rows >> 16) & 0xFFFF) | ((cols & 0xFFFF) << 16);
  g1[4] = (rows & 0xFFFF);
  g1[5] = src_stride_elems;
  g1[6] = 0; g1[7] = 0;
  i32x4 g2 = {0, 0, 0, 0};
  i32x4 g3 = {0, 0, 0, 0};
#if __has_include(<hip/amd_detail/amd_gfx1250_TDM.h>)
  i32x8 g4 = {0, 0, 0, 0, 0, 0, 0, 0};
  __builtin_amdgcn_tensor_load_to_lds(g0, g1, g2, g3, g4, 0);
#else
  __builtin_amdgcn_tensor_load_to_lds(g0, g1, g2, g3, 0);
#endif
}
#endif

// ---------------------------------------------------------------------------
// WMMA GEMM core: 128 threads = 4 waves, 32(M) x 64(N) macro tile, K step 32.
// LDS tiles in WMMA fragment order; each wave -> one A frag, two B frags,
// two v_wmma per K step. Wt: bf16, transposed [N][K].
// ---------------------------------------------------------------------------
template <typename AT>
DEVI void gemm_core(const AT* __restrict__ A, int lda,
                    const float* __restrict__ tailA,
                    const __bf16* __restrict__ Wt, int Kld, int K,
                    int m0, int n0, __bf16* lA, __bf16* lB, v8f* acc)
{
  const int tid  = threadIdx.x;
  const int lane = tid & 31;
  const int wave = tid >> 5;
  const int wmt = wave >> 1;
  const int wnt = wave & 1;
  const int sr  = tid >> 2;               // A stage row 0..31
  const int sc0 = (tid & 3) << 3;         // A stage col base 0,8,16,24
  for (int k0 = 0; k0 < K; k0 += 32) {
    // --- stage A tile (32 x 32) into fragment order ---
    {
      v8bf pk;
      long ar = (long)(m0 + sr);
      int kk0 = k0 + sc0;
      if (tailA && kk0 >= CA) {
        __bf16 tv = f2bf(tailA[ar]);
        #pragma unroll
        for (int i = 0; i < 8; ++i) pk[i] = tv;
      } else {
        pk = load8(&A[ar * (long)lda + kk0]);
      }
      int dst = ((sr >> 4) << 9) + (((sr & 15) + (((sc0 >> 3) & 1) << 4)) << 4)
              + ((sc0 >> 4) << 3);
      *(v8bf*)&lA[dst] = pk;
    }
    // --- stage B tile (64 n x 32 k) from transposed bf16 weights ---
    #pragma unroll
    for (int t = 0; t < 2; ++t) {
      int e = (tid + t * 128) * 8;
      int nl = e >> 5, kc = e & 31;        // n-local 0..63, k-chunk 0,8,16,24
      v8bf pk = *(const v8bf*)&Wt[(long)(n0 + nl) * Kld + k0 + kc];
      int dst = ((nl >> 4) << 9) + (((nl & 15) + (((kc >> 3) & 1) << 4)) << 4)
              + ((kc >> 4) << 3);
      *(v8bf*)&lB[dst] = pk;
    }
    if (k0 + 32 < K) {                     // gfx1250 global_prefetch path
      if (!tailA || (k0 + 32) < CA)
        __builtin_prefetch(&A[(long)(m0 + sr) * lda + k0 + 32], 0, 1);
      __builtin_prefetch(&Wt[(long)(n0 + (tid >> 1)) * Kld + k0 + 32], 0, 1);
    }
    __syncthreads();
    v16bf af = frag16(&lA[(wmt << 9) + (lane << 4)]);
    v16bf b0 = frag16(&lB[(wnt << 9) + (lane << 4)]);
    v16bf b1 = frag16(&lB[((wnt + 2) << 9) + (lane << 4)]);
    acc[0] = bwmma(af, b0, acc[0]);
    acc[1] = bwmma(af, b1, acc[1]);
    __syncthreads();
  }
}

#define TILE_EPI \
  const int lane = threadIdx.x & 31; \
  const int wave = threadIdx.x >> 5; \
  const int wm = (wave >> 1) << 4; \
  const int wn = (wave & 1) << 4; \
  const int hi = lane >> 4, lo = lane & 15;

// ---------------- K0: weight prep  dst[n][k] = bf16(src[k][n]) ----------------
__global__ __launch_bounds__(256)
void k_prep_wt(const float* __restrict__ src, __bf16* __restrict__ dst,
               int K, int N)
{
  int idx = blockIdx.x * 256 + threadIdx.x;
  if (idx >= K * N) return;
  int k = idx / N, n = idx - k * N;
  dst[(long)n * K + k] = f2bf(src[idx]);
}

// ---------------- K1: qtok = a @ Wa ----------------
__global__ __launch_bounds__(128)
void k_gemm_qtok(const float* __restrict__ a, const __bf16* __restrict__ WaT,
                 float* __restrict__ qtok)
{
  __shared__ v16bf lAv[64], lBv[128];
  int m0 = blockIdx.x * 32, n0 = blockIdx.y * 64;
  v8f acc[2] = {{0.f,0.f,0.f,0.f,0.f,0.f,0.f,0.f},
                {0.f,0.f,0.f,0.f,0.f,0.f,0.f,0.f}};
  gemm_core(a, CT, nullptr, WaT, CT, CT, m0, n0, (__bf16*)lAv, (__bf16*)lBv, acc);
  TILE_EPI
  #pragma unroll
  for (int g = 0; g < 2; ++g)
    #pragma unroll
    for (int r = 0; r < 8; ++r) {
      int m = m0 + wm + hi * 8 + r, n = n0 + wn + lo + g * 32;
      qtok[(long)m * CA + n] = acc[g][r];
    }
}

// ---------------- K2: gather + mask + padded-LN of extra_feats ----------------
__global__ __launch_bounds__(128)
void k_setup(const float* __restrict__ qtok, const float* __restrict__ ef,
             const float* __restrict__ amask, const int* __restrict__ a2t,
             float* __restrict__ x, __bf16* __restrict__ sn,
             float* __restrict__ tail)
{
  __shared__ float sbuf[4];
  long row = blockIdx.x;
  int c = threadIdx.x;
  int b = (int)(row >> 14);
  float e = ef[row * CA + c];
  int tok = a2t[row];
  float qv = qtok[((long)b * NTOK + tok) * CA + c];
  float m = amask[row];
  x[row * CA + c] = (qv + e) * m;
  float sum = block_sum_128(e, sbuf);
  float mu = sum * (1.f / (float)CS);
  float d = e - mu;
  float ssq = block_sum_128(d * d, sbuf);
  float var = (ssq + (float)(CS - CA) * mu * mu) * (1.f / (float)CS);
  float rs = rsqrtf(var + 1e-5f);
  sn[row * CA + c] = f2bf(d * rs);
  if (c == 0) tail[row] = -mu * rs;
}

// ------- K3: conditioning GEMM, six fused weight segments (N=768) -------
__global__ __launch_bounds__(128)
void k_gemm_cond(const __bf16* __restrict__ sn, const float* __restrict__ tail,
                 const __bf16* __restrict__ W0, const __bf16* __restrict__ W1,
                 const __bf16* __restrict__ W2, const __bf16* __restrict__ W3,
                 const __bf16* __restrict__ W4, const __bf16* __restrict__ W5,
                 const float* __restrict__ B0, const float* __restrict__ B2,
                 const float* __restrict__ B3, const float* __restrict__ B5,
                 float* __restrict__ O0, float* __restrict__ O1,
                 float* __restrict__ O2, float* __restrict__ O3,
                 float* __restrict__ O4, float* __restrict__ O5)
{
  __shared__ v16bf lAv[64], lBv[128];
  int m0 = blockIdx.x * 32;
  int nglob = blockIdx.y * 64;               // 0..767
  int widx = nglob >> 7;
  int n0 = nglob & 127;                      // 0 or 64
  const __bf16* Ws[6] = {W0, W1, W2, W3, W4, W5};
  const float* Bs[6] = {B0, nullptr, B2, B3, nullptr, B5};
  float* Os[6] = {O0, O1, O2, O3, O4, O5};
  const bool sig[6] = {true, false, true, true, false, true};
  v8f acc[2] = {{0.f,0.f,0.f,0.f,0.f,0.f,0.f,0.f},
                {0.f,0.f,0.f,0.f,0.f,0.f,0.f,0.f}};
  gemm_core(sn, CA, tail, Ws[widx], CS, CS, m0, n0,
            (__bf16*)lAv, (__bf16*)lBv, acc);
  TILE_EPI
  const float* bias = Bs[widx];
  float* out = Os[widx];
  #pragma unroll
  for (int g = 0; g < 2; ++g)
    #pragma unroll
    for (int r = 0; r < 8; ++r) {
      int m = m0 + wm + hi * 8 + r, n = n0 + wn + lo + g * 32;
      float v = acc[g][r] + (bias ? bias[n] : 0.f);
      if (sig[widx]) v = 1.f / (1.f + __expf(-v));
      out[(long)m * CA + n] = v;
    }
}

// ---------------- K4/K8: AdaLN -> bf16 ----------------
__global__ __launch_bounds__(128)
void k_adaln(const float* __restrict__ x, const float* __restrict__ cgate,
             const float* __restrict__ cbias, __bf16* __restrict__ out)
{
  __shared__ float sbuf[4];
  long row = blockIdx.x; int c = threadIdx.x;
  float v = x[row * CA + c];
  float mu = block_sum_128(v, sbuf) * (1.f / CA);
  float d = v - mu;
  float var = block_sum_128(d * d, sbuf) * (1.f / CA);
  float xn = d * rsqrtf(var + 1e-5f);
  out[row * CA + c] = f2bf(cgate[row * CA + c] * xn + cbias[row * CA + c]);
}

// ------- K5: fused QKV+gate GEMM; q/k/v bf16, gate sigmoid f32 -------
__global__ __launch_bounds__(128)
void k_gemm_qkvg(const __bf16* __restrict__ xa,
                 const __bf16* __restrict__ Wq, const __bf16* __restrict__ Wk,
                 const __bf16* __restrict__ Wv, const __bf16* __restrict__ Wg,
                 const float* __restrict__ bq,
                 __bf16* __restrict__ Oq, __bf16* __restrict__ Ok,
                 __bf16* __restrict__ Ov, float* __restrict__ Og)
{
  __shared__ v16bf lAv[64], lBv[128];
  int m0 = blockIdx.x * 32, nglob = blockIdx.y * 64;
  int widx = nglob >> 7, n0 = nglob & 127;
  const __bf16* Ws[4] = {Wq, Wk, Wv, Wg};
  __bf16* Ob[3] = {Oq, Ok, Ov};
  v8f acc[2] = {{0.f,0.f,0.f,0.f,0.f,0.f,0.f,0.f},
                {0.f,0.f,0.f,0.f,0.f,0.f,0.f,0.f}};
  gemm_core(xa, CA, nullptr, Ws[widx], CA, CA, m0, n0,
            (__bf16*)lAv, (__bf16*)lBv, acc);
  TILE_EPI
  #pragma unroll
  for (int g = 0; g < 2; ++g)
    #pragma unroll
    for (int r = 0; r < 8; ++r) {
      int m = m0 + wm + hi * 8 + r, n = n0 + wn + lo + g * 32;
      float v = acc[g][r];
      if (widx == 0) v += bq[n];
      if (widx == 3) {
        Og[(long)m * CA + n] = 1.f / (1.f + __expf(-v));
      } else {
        Ob[widx][(long)m * CA + n] = f2bf(v);
      }
    }
}

// --------- K6: sliding-window attention; out = bf16(gate * o) ---------
__global__ __launch_bounds__(256)
void k_attn(const __bf16* __restrict__ qb, const __bf16* __restrict__ kb,
            const __bf16* __restrict__ vb, const float* __restrict__ plm,
            const float* __restrict__ pg, const float* __restrict__ pb,
            const float* __restrict__ wpb, const float* __restrict__ amask,
            const float* __restrict__ gh, __bf16* __restrict__ ob)
{
  extern __shared__ char smem[];
  __bf16* lq  = (__bf16*)smem;               // [32][136]
  __bf16* lk  = lq + 32 * 136;               // [128][136]  row=key
  __bf16* lvT = lk + 128 * 136;              // [128][136]  row=dh (transposed)
  float*  lsc = (float*)(lvT + 128 * 136);   // [h][32][128]
  float*  lkm = lsc + NH * 32 * 128;
  float*  spg = lkm + 128;
  float*  spb = spg + 16;
  float*  swp = spb + 16;

  const int tid = threadIdx.x;
  const int bw = blockIdx.x;
  const int b = bw >> 9, w = bw & 511;
  const long qrow0 = (long)b * NATOM + (long)w * NQW;
  const int kbase = w * NQW + NQW / 2 - NKW / 2;
  const bool interior = (w >= 2) && (w < NW - 2);

  if (tid < 16) { spg[tid] = pg[tid]; spb[tid] = pb[tid]; }
  if (tid >= 64 && tid < 128) swp[tid - 64] = wpb[tid - 64];

  bool tdm_q = false, tdm_k = false;
#if TDM_OK
  tdm_q = true;
  tdm_k = interior;
  if ((tid >> 5) == 0) {
    unsigned base = (unsigned)__builtin_amdgcn_groupstaticsize();
    tdm_load_2d_bf16(qb + qrow0 * CA,
                     base + (unsigned)((char*)lq - (char*)smem), NQW, CA, CA);
    if (tdm_k)
      tdm_load_2d_bf16(kb + ((long)b * NATOM + kbase) * CA,
                       base + (unsigned)((char*)lk - (char*)smem), NKW, CA, CA);
  }
#endif
  if (!tdm_q) {
    for (int ch = tid; ch < NQW * CA / 8; ch += 256) {
      int r = (ch * 8) >> 7, c0 = (ch * 8) & 127;
      *(v8bf*)&lq[r * 136 + c0] = *(const v8bf*)&qb[(qrow0 + r) * CA + c0];
    }
  }
  if (!tdm_k) {
    for (int ch = tid; ch < NKW * CA / 8; ch += 256) {
      int r = (ch * 8) >> 7, c0 = (ch * 8) & 127;
      int cl = min(max(kbase + r, 0), NATOM - 1);
      *(v8bf*)&lk[r * 136 + c0] =
          *(const v8bf*)&kb[((long)b * NATOM + cl) * CA + c0];
    }
  }
  for (int ch = tid; ch < NKW * CA / 8; ch += 256) {
    int r = (ch * 8) >> 7, c0 = (ch * 8) & 127;
    int cl = min(max(kbase + r, 0), NATOM - 1);
    v8bf vv = *(const v8bf*)&vb[((long)b * NATOM + cl) * CA + c0];
    #pragma unroll
    for (int i = 0; i < 8; ++i) lvT[(c0 + i) * 136 + r] = vv[i];
  }
  if (tid < 128) {
    int aidx = kbase + tid;
    int cl = min(max(aidx, 0), NATOM - 1);
    float valid = (aidx >= 0 && aidx < NATOM) ? 1.f : 0.f;
    lkm[tid] = valid * amask[(long)b * NATOM + cl];
  }
#if TDM_OK
  if ((tdm_q || tdm_k) && (tid >> 5) == 0)
    __builtin_amdgcn_s_wait_tensorcnt(0);
#endif
  __syncthreads();

  // pair bias: (LN(p)*pg+pb) @ wpb -> lsc[h][q][k]
  const float* pw = plm + (long)bw * NQW * NKW * CP;
  for (int pr = tid; pr < NQW * NKW; pr += 256) {
    const float* pp = pw + (long)pr * CP;
    float pv[16]; float s = 0.f;
    #pragma unroll
    for (int c = 0; c < 16; ++c) { pv[c] = pp[c]; s += pv[c]; }
    float mu = s * (1.f / 16.f); float s2 = 0.f;
    #pragma unroll
    for (int c = 0; c < 16; ++c) { float d = pv[c] - mu; s2 += d * d; }
    float rs = rsqrtf(s2 * (1.f / 16.f) + 1e-5f);
    float bh[4] = {0.f, 0.f, 0.f, 0.f};
    #pragma unroll
    for (int c = 0; c < 16; ++c) {
      float pn = (pv[c] - mu) * rs * spg[c] + spb[c];
      #pragma unroll
      for (int h = 0; h < 4; ++h) bh[h] += pn * swp[c * 4 + h];
    }
    #pragma unroll
    for (int h = 0; h < 4; ++h) lsc[(h * 32 * 128) + pr] = bh[h];
  }
  __syncthreads();

  const int lane = tid & 31, wave = tid >> 5;
  const int hi = lane >> 4, lo = lane & 15;
  const float inv = 0.17677669529663687f;

  for (int t = wave; t < 64; t += 8) {
    int h = t >> 4, mt = (t >> 3) & 1, nt = t & 7;
    v16bf af = frag16_gap(&lq[(mt * 16 + lo) * 136 + h * 32 + hi * 8]);
    v16bf bfv = frag16_gap(&lk[(nt * 16 + lo) * 136 + h * 32 + hi * 8]);
    v8f acc = {0.f,0.f,0.f,0.f,0.f,0.f,0.f,0.f};
    acc = bwmma(af, bfv, acc);
    #pragma unroll
    for (int r = 0; r < 8; ++r) {
      int m = mt * 16 + hi * 8 + r, n = nt * 16 + lo;
      int o = (h * 32 + m) * 128 + n;
      float sc = acc[r] * inv + lsc[o];
      lsc[o] = (lkm[n] > 0.f) ? sc : -1e9f;
    }
  }
  __syncthreads();

  for (int rr = wave * 16; rr < wave * 16 + 16; ++rr) {
    int h = rr >> 5, q = rr & 31;
    float* rowp = lsc + (h * 32 + q) * 128;
    float vals[4]; float mx = -1e30f;
    #pragma unroll
    for (int s = 0; s < 4; ++s) {
      vals[s] = rowp[lane + 32 * s];
      mx = fmaxf(mx, vals[s]);
    }
    mx = wave_max(mx);
    float sum = 0.f;
    #pragma unroll
    for (int s = 0; s < 4; ++s) { vals[s] = __expf(vals[s] - mx); sum += vals[s]; }
    sum = wave_sum(sum);
    float rcp = 1.f / sum;
    #pragma unroll
    for (int s = 0; s < 4; ++s) rowp[lane + 32 * s] = vals[s] * rcp;
  }
  __syncthreads();

  for (int t = wave; t < 16; t += 8) {
    int h = t >> 2, mt = (t >> 1) & 1, nt = t & 1;
    v8f acc = {0.f,0.f,0.f,0.f,0.f,0.f,0.f,0.f};
    for (int k0 = 0; k0 < 128; k0 += 32) {
      const float* ap = lsc + (h * 32 + mt * 16 + lo) * 128 + k0 + hi * 8;
      v8f f0 = *(const v8f*)ap;
      v8f f1 = *(const v8f*)(ap + 16);
      v16bf af;
      #pragma unroll
      for (int j = 0; j < 8; ++j) { af[j] = f2bf(f0[j]); af[j + 8] = f2bf(f1[j]); }
      v16bf bfv = frag16_gap(&lvT[(h * 32 + nt * 16 + lo) * 136 + k0 + hi * 8]);
      acc = bwmma(af, bfv, acc);
    }
    #pragma unroll
    for (int r = 0; r < 8; ++r) {
      int m = mt * 16 + hi * 8 + r, n = nt * 16 + lo;
      long off = (qrow0 + m) * CA + h * 32 + n;
      ob[off] = f2bf(acc[r] * gh[off]);        // fuse output gate here
    }
  }
}

// ------- K7/K10: residual GEMM  x += scale * (A @ W) -------
__global__ __launch_bounds__(128)
void k_gemm_resid(const __bf16* __restrict__ Ain, const __bf16* __restrict__ Wt,
                  int K, const float* __restrict__ scale, float* __restrict__ x)
{
  __shared__ v16bf lAv[64], lBv[128];
  int m0 = blockIdx.x * 32, n0 = blockIdx.y * 64;
  v8f acc[2] = {{0.f,0.f,0.f,0.f,0.f,0.f,0.f,0.f},
                {0.f,0.f,0.f,0.f,0.f,0.f,0.f,0.f}};
  gemm_core(Ain, K, nullptr, Wt, K, K, m0, n0, (__bf16*)lAv, (__bf16*)lBv, acc);
  TILE_EPI
  #pragma unroll
  for (int g = 0; g < 2; ++g)
    #pragma unroll
    for (int r = 0; r < 8; ++r) {
      int m = m0 + wm + hi * 8 + r, n = n0 + wn + lo + g * 32;
      long off = (long)m * CA + n;
      x[off] += scale[off] * acc[g][r];
    }
}

// ---------------- K9: SwiGLU -> bf16 ----------------
__global__ __launch_bounds__(128)
void k_gemm_swiglu(const __bf16* __restrict__ xt,
                   const __bf16* __restrict__ W1t, const __bf16* __restrict__ W2t,
                   __bf16* __restrict__ hsw)
{
  __shared__ v16bf lAv[64], lBv[128];
  int m0 = blockIdx.x * 32, n0 = blockIdx.y * 64;
  v8f a1[2] = {{0.f,0.f,0.f,0.f,0.f,0.f,0.f,0.f},
               {0.f,0.f,0.f,0.f,0.f,0.f,0.f,0.f}};
  v8f a2[2] = {{0.f,0.f,0.f,0.f,0.f,0.f,0.f,0.f},
               {0.f,0.f,0.f,0.f,0.f,0.f,0.f,0.f}};
  gemm_core(xt, CA, nullptr, W1t, CA, CA, m0, n0, (__bf16*)lAv, (__bf16*)lBv, a1);
  gemm_core(xt, CA, nullptr, W2t, CA, CA, m0, n0, (__bf16*)lAv, (__bf16*)lBv, a2);
  TILE_EPI
  #pragma unroll
  for (int g = 0; g < 2; ++g)
    #pragma unroll
    for (int r = 0; r < 8; ++r) {
      int m = m0 + wm + hi * 8 + r, n = n0 + wn + lo + g * 32;
      float u = a1[g][r];
      float sg = u / (1.f + __expf(-u));
      hsw[(long)m * 2 * CA + n] = f2bf(sg * a2[g][r]);
    }
}

// ---------------- K11: final LN + 128->3 projection ----------------
__global__ __launch_bounds__(128)
void k_final(const float* __restrict__ x, const float* __restrict__ amask,
             const float* __restrict__ g, const float* __restrict__ bl,
             const float* __restrict__ Wout, float* __restrict__ out)
{
  __shared__ float sbuf[4];
  __shared__ float racc[12];
  long row = blockIdx.x; int c = threadIdx.x;
  float v = x[row * CA + c] * amask[row];
  float mu = block_sum_128(v, sbuf) * (1.f / CA);
  float d = v - mu;
  float var = block_sum_128(d * d, sbuf) * (1.f / CA);
  float xn = d * rsqrtf(var + 1e-5f) * g[c] + bl[c];
  int wv = threadIdx.x >> 5;
  #pragma unroll
  for (int j = 0; j < 3; ++j) {
    float pj = wave_sum(xn * Wout[c * 3 + j]);
    if ((threadIdx.x & 31) == 0) racc[j * 4 + wv] = pj;
  }
  __syncthreads();
  if (c < 3)
    out[row * 3 + c] = racc[c * 4] + racc[c * 4 + 1] + racc[c * 4 + 2] + racc[c * 4 + 3];
}

// ---------------------------------------------------------------------------
extern "C" void kernel_launch(void* const* d_in, const int* in_sizes, int n_in,
                              void* d_out, int out_size, void* d_ws, size_t ws_size,
                              hipStream_t stream) {
  (void)in_sizes; (void)n_in; (void)out_size; (void)ws_size;
  const float* a     = (const float*)d_in[0];
  const float* ef    = (const float*)d_in[2];
  const float* plm   = (const float*)d_in[3];
  const float* am    = (const float*)d_in[4];
  const int*   a2t   = (const int*)d_in[5];
  const float* Wa    = (const float*)d_in[6];
  const float* lnq_g = (const float*)d_in[7];
  const float* lnq_b = (const float*)d_in[8];
  const float* Wout  = (const float*)d_in[9];
  const float* ag_w  = (const float*)d_in[10];
  const float* ag_b  = (const float*)d_in[11];
  const float* ab_w  = (const float*)d_in[12];
  const float* wq    = (const float*)d_in[13];
  const float* bq    = (const float*)d_in[14];
  const float* wk    = (const float*)d_in[15];
  const float* wv    = (const float*)d_in[16];
  const float* pg    = (const float*)d_in[17];
  const float* pb    = (const float*)d_in[18];
  const float* wpb   = (const float*)d_in[19];
  const float* wg    = (const float*)d_in[20];
  const float* wo    = (const float*)d_in[21];
  const float* sk_w  = (const float*)d_in[22];
  const float* sk_b  = (const float*)d_in[23];
  const float* tg_w  = (const float*)d_in[24];
  const float* tg_b  = (const float*)d_in[25];
  const float* tb_w  = (const float*)d_in[26];
  const float* wt1   = (const float*)d_in[27];
  const float* wt2   = (const float*)d_in[28];
  const float* wto   = (const float*)d_in[29];
  const float* tk_w  = (const float*)d_in[30];
  const float* tk_b  = (const float*)d_in[31];

  char* wsb = (char*)d_ws;
  size_t off = 0;
  auto alloc = [&](size_t bytes) -> void* {
    void* p = (void*)(wsb + off);
    off += (bytes + 255) & ~(size_t)255;
    return p;
  };
  const size_t RC = (size_t)MROWS * CA * sizeof(float);    // 16 MB
  float*  qtok = (float*)alloc((size_t)BB * NTOK * CA * sizeof(float));
  float*  x    = (float*)alloc(RC);
  __bf16* sn   = (__bf16*)alloc(RC / 2);
  float*  tail = (float*)alloc((size_t)MROWS * sizeof(float));
  float* c_ag = (float*)alloc(RC); float* c_ab = (float*)alloc(RC);
  float* c_sk = (float*)alloc(RC); float* c_tg = (float*)alloc(RC);
  float* c_tb = (float*)alloc(RC); float* c_tk = (float*)alloc(RC);
  __bf16* xa  = (__bf16*)alloc(RC / 2);
  __bf16* qh  = (__bf16*)alloc(RC / 2);
  __bf16* kh  = (__bf16*)alloc(RC / 2);
  __bf16* vh  = (__bf16*)alloc(RC / 2);
  float*  gh  = (float*)alloc(RC);
  __bf16* obuf = (__bf16*)alloc(RC / 2);
  __bf16* hsw  = (__bf16*)alloc(RC);                        // 32768 x 256 bf16

  // bf16 transposed weights
  auto walloc = [&](size_t elems) -> __bf16* {
    return (__bf16*)alloc(elems * sizeof(__bf16));
  };
  __bf16* WaT = walloc(CT * CA);
  __bf16 *agT[NBLK], *abT[NBLK], *skT[NBLK], *tgT[NBLK], *tbT[NBLK], *tkT[NBLK];
  __bf16 *wqT[NBLK], *wkT[NBLK], *wvT[NBLK], *wgT[NBLK], *woT[NBLK];
  __bf16 *wt1T[NBLK], *wt2T[NBLK], *wtoT[NBLK];
  for (int i = 0; i < NBLK; ++i) {
    agT[i] = walloc(CS * CA);  abT[i] = walloc(CS * CA);
    skT[i] = walloc(CS * CA);  tgT[i] = walloc(CS * CA);
    tbT[i] = walloc(CS * CA);  tkT[i] = walloc(CS * CA);
    wqT[i] = walloc(CA * CA);  wkT[i] = walloc(CA * CA);
    wvT[i] = walloc(CA * CA);  wgT[i] = walloc(CA * CA);
    woT[i] = walloc(CA * CA);
    wt1T[i] = walloc(CA * 2 * CA); wt2T[i] = walloc(CA * 2 * CA);
    wtoT[i] = walloc(2 * CA * CA);
  }

  auto prep = [&](const float* src, __bf16* dst, int K, int N) {
    int tot = K * N;
    k_prep_wt<<<(tot + 255) / 256, 256, 0, stream>>>(src, dst, K, N);
  };
  prep(Wa, WaT, CT, CA);
  for (int i = 0; i < NBLK; ++i) {
    const long wS = (long)i * CS * CA;
    const long wA = (long)i * CA * CA;
    const long wT = (long)i * CA * 2 * CA;
    prep(ag_w + wS, agT[i], CS, CA);  prep(ab_w + wS, abT[i], CS, CA);
    prep(sk_w + wS, skT[i], CS, CA);  prep(tg_w + wS, tgT[i], CS, CA);
    prep(tb_w + wS, tbT[i], CS, CA);  prep(tk_w + wS, tkT[i], CS, CA);
    prep(wq + wA, wqT[i], CA, CA);    prep(wk + wA, wkT[i], CA, CA);
    prep(wv + wA, wvT[i], CA, CA);    prep(wg + wA, wgT[i], CA, CA);
    prep(wo + wA, woT[i], CA, CA);
    prep(wt1 + wT, wt1T[i], CA, 2 * CA);
    prep(wt2 + wT, wt2T[i], CA, 2 * CA);
    prep(wto + wT, wtoT[i], 2 * CA, CA);
  }

  const size_t attn_lds =
      (size_t)(32 + 128 + 128) * 136 * sizeof(__bf16) +
      (size_t)NH * 32 * 128 * sizeof(float) +
      (128 + 16 + 16 + 64) * sizeof(float);

  dim3 blk128(128);
  k_gemm_qtok<<<dim3((BB * NTOK) / 32, CA / 64), blk128, 0, stream>>>(a, WaT, qtok);
  k_setup<<<MROWS, 128, 0, stream>>>(qtok, ef, am, a2t, x, sn, tail);

  for (int i = 0; i < NBLK; ++i) {
    k_gemm_cond<<<dim3(MROWS / 32, (6 * CA) / 64), blk128, 0, stream>>>(
        sn, tail, agT[i], abT[i], skT[i], tgT[i], tbT[i], tkT[i],
        ag_b + (long)i * CA, sk_b + (long)i * CA,
        tg_b + (long)i * CA, tk_b + (long)i * CA,
        c_ag, c_ab, c_sk, c_tg, c_tb, c_tk);
    k_adaln<<<MROWS, 128, 0, stream>>>(x, c_ag, c_ab, xa);
    k_gemm_qkvg<<<dim3(MROWS / 32, (4 * CA) / 64), blk128, 0, stream>>>(
        xa, wqT[i], wkT[i], wvT[i], wgT[i], bq + (long)i * CA,
        qh, kh, vh, gh);
    k_attn<<<BB * NW, 256, attn_lds, stream>>>(
        qh, kh, vh, plm, pg + (long)i * CP, pb + (long)i * CP,
        wpb + (long)i * CP * NH, am, gh, obuf);
    k_gemm_resid<<<dim3(MROWS / 32, CA / 64), blk128, 0, stream>>>(
        obuf, woT[i], CA, c_sk, x);
    k_adaln<<<MROWS, 128, 0, stream>>>(x, c_tg, c_tb, xa);
    k_gemm_swiglu<<<dim3(MROWS / 32, (2 * CA) / 64), blk128, 0, stream>>>(
        xa, wt1T[i], wt2T[i], hsw);
    k_gemm_resid<<<dim3(MROWS / 32, CA / 64), blk128, 0, stream>>>(
        hsw, wtoT[i], 2 * CA, c_tk, x);
  }
  k_final<<<MROWS, 128, 0, stream>>>(x, am, lnq_g, lnq_b, Wout, (float*)d_out);
}
